// ModulatedConv2d_39573828665625
// MI455X (gfx1250) — matrix operand order repaired and checked by reading, not compiled, not run on hardware
//
#include <hip/hip_runtime.h>
#include <hip/hip_bf16.h>
#include <math.h>

typedef __attribute__((ext_vector_type(16))) __bf16 v16bf;
typedef __attribute__((ext_vector_type(8)))  float  v8f;

constexpr int NB   = 8;     // batch
constexpr int CIN  = 512;
constexpr int COUT = 256;
constexpr int HH   = 64;
constexpr int WW   = 64;
constexpr int ZZ   = 512;
constexpr int OH   = 129;   // conv_transpose output spatial
constexpr int FH   = 128;   // final output spatial
constexpr int XW   = 34;    // staged x-window (32 pixels + 2 halo)

static __device__ __forceinline__ unsigned short f2bf(float f) {
    union { float f; unsigned int u; } v; v.f = f;
    unsigned int u = v.u;
    return (unsigned short)((u + 0x7FFFu + ((u >> 16) & 1u)) >> 16);
}

// ---------------------------------------------------------------------------
// 1) modulation: s[b][ci] = dot(style[b], mod_weight[ci]) / sqrt(Z) + bias[ci]
// ---------------------------------------------------------------------------
__global__ void mod_kernel(const float* __restrict__ style,
                           const float* __restrict__ modw,
                           const float* __restrict__ modb,
                           float* __restrict__ s) {
    int gid = blockIdx.x * blockDim.x + threadIdx.x;
    if (gid >= NB * CIN) return;
    int b = gid / CIN, ci = gid % CIN;
    const float* st = style + (size_t)b * ZZ;
    const float* mw = modw + (size_t)ci * ZZ;
    float acc = 0.f;
    for (int z = 0; z < ZZ; z += 4) {
        acc += st[z] * mw[z] + st[z + 1] * mw[z + 1]
             + st[z + 2] * mw[z + 2] + st[z + 3] * mw[z + 3];
    }
    s[gid] = acc * 0.04419417382415922f /* 1/sqrt(512) */ + modb[ci];
}

// ---------------------------------------------------------------------------
// 2) weight prep: w = scale*weight*s ; demod ; convert to bf16 and scatter
//    into WMMA 16x32 A-fragment layout (frag = 32 lanes x 16 bf16 = 1024B)
//    frag index: ((b*9 + tap)*16 + co/16)*16 + ci/32
// ---------------------------------------------------------------------------
__global__ __launch_bounds__(256)
void wprep_kernel(const float* __restrict__ weight,
                  const float* __restrict__ s,
                  unsigned short* __restrict__ aw) {
    const int b   = blockIdx.x / COUT;
    const int co  = blockIdx.x % COUT;
    const int tid = threadIdx.x;
    const float scale = 0.014731391274719739f; // 1/sqrt(512*9)

    const float* wb = weight + (size_t)co * CIN * 9;
    const float* sb = s + (size_t)b * CIN;

    float wv[18];
    float ss = 0.f;
    for (int j = 0; j < 18; ++j) {
        int e = tid * 18 + j;          // 0..4607 over (ci,tap)
        int ci = e / 9, tap = e % 9;
        float w = scale * wb[ci * 9 + tap] * sb[ci];
        wv[j] = w;
        ss += w * w;
    }
    __shared__ float red[256];
    red[tid] = ss;
    __syncthreads();
    for (int off = 128; off > 0; off >>= 1) {
        if (tid < off) red[tid] += red[tid + off];
        __syncthreads();
    }
    const float demod = rsqrtf(red[0] + 1e-8f);

    for (int j = 0; j < 18; ++j) {
        int e = tid * 18 + j;
        int ci = e / 9, tap = e % 9;
        int m = co & 15, k = ci & 31;
        // A-matrix 16x32 bf16 layout used here (contiguous k, same order as B):
        // lanes 0-15 : M=lane, K = 0..15
        // lanes 16-31: M=lane-16, K = 16..31
        int lane, slot;
        if (k < 16)      { lane = m;      slot = k;      }
        else             { lane = m + 16; slot = k - 16; }
        size_t frag = ((size_t)(b * 9 + tap) * 16 + (co >> 4)) * 16 + (ci >> 5);
        aw[frag * 512 + lane * 16 + slot] = f2bf(wv[j] * demod);
    }
}

// ---------------------------------------------------------------------------
// 3) conv_transpose(stride 2) as bf16 WMMA implicit GEMM over parity classes.
//    Block: (xg, oy, b*2+ch), 256 threads (8 waves). Each wave owns 16 Cout
//    and TWO 16-pixel N-tiles (32 same-parity x positions) -> each A fragment
//    feeds two WMMAs, halving A-fragment (L2) traffic.
// ---------------------------------------------------------------------------
__global__ __launch_bounds__(256)
void convt_wmma_kernel(const float* __restrict__ in,
                       const unsigned short* __restrict__ aw,
                       float* __restrict__ convo) {
    const int xg  = blockIdx.x;        // 0..4 (0..2 even-x groups, 3..4 odd-x)
    const int oy  = blockIdx.y;        // 0..128
    const int bz  = blockIdx.z;        // b*2 + coHalf
    const int b   = bz >> 1;
    const int ch  = bz & 1;
    const int tid = threadIdx.x;
    const int lane = tid & 31;
    const int wid  = tid >> 5;
    const int nL   = lane & 15;

    int px, g;
    if (xg < 3) { px = 0; g = xg; } else { px = 1; g = xg - 3; }

    // valid (ky, iy) taps for this output row: oy = 2*iy + ky
    int nY = 0, iyA[2], kyA[2];
    if ((oy & 1) == 0) {
        int t = oy >> 1;
        if (t < HH)     { iyA[nY] = t;     kyA[nY] = 0; nY++; }
        if (t - 1 >= 0) { iyA[nY] = t - 1; kyA[nY] = 2; nY++; }
    } else {
        iyA[0] = (oy - 1) >> 1; kyA[0] = 1; nY = 1;
    }
    // valid (kx, lds x-offset) taps for this x parity
    int nX, kxA[2], xoffA[2];
    if (px == 0) { nX = 2; kxA[0] = 0; xoffA[0] = 1; kxA[1] = 2; xoffA[1] = 0; }
    else         { nX = 1; kxA[0] = 1; xoffA[0] = 1; }

    // Stage up to 2 input rows as bf16: window ix in [g*32-1, g*32+32]
    // LDS layout [row][xl][ci] -> B fragments are contiguous per lane.
    __shared__ __align__(64) unsigned short smem[2 * XW * CIN];
    const int iw0 = g * 32 - 1;
    for (int yi = 0; yi < 2; ++yi) {
        const int rowValid = (yi < nY);
        const int iy = rowValid ? iyA[yi] : 0;
        const float* src = in + ((size_t)b * CIN * HH + iy) * WW;
        for (int e = tid; e < XW * CIN; e += 256) {
            int ci = e / XW;
            int xl = e % XW;
            int ix = iw0 + xl;
            float v = 0.f;
            if (rowValid && ix >= 0 && ix < WW)
                v = src[(size_t)ci * HH * WW + ix];
            smem[yi * (XW * CIN) + xl * CIN + ci] = f2bf(v);
        }
    }
    __syncthreads();

    const int coT = ch * 8 + wid;                  // 16-wide Cout tile index
    v8f acc0 = {0.f, 0.f, 0.f, 0.f, 0.f, 0.f, 0.f, 0.f};
    v8f acc1 = {0.f, 0.f, 0.f, 0.f, 0.f, 0.f, 0.f, 0.f};

    for (int yi = 0; yi < nY; ++yi) {
        for (int xi = 0; xi < nX; ++xi) {
            const int tap  = kyA[yi] * 3 + kxA[xi];
            const int xoff = xoffA[xi];
            const unsigned short* afrag =
                aw + ((size_t)((b * 9 + tap) * 16 + coT)) * 16 * 512 + lane * 16;
            // B 32x16 bf16: lanes 0-15 -> n=lane, K 0..15; lanes 16-31 -> K 16..31
            const int kSel  = (lane < 16) ? 0 : 16;
            const int bCol0 = yi * (XW * CIN) + (xoff + nL) * CIN + kSel;
            const int bCol1 = bCol0 + 16 * CIN;    // second 16-pixel sub-tile
            for (int cc = 0; cc < 16; ++cc) {
                v16bf a  = *(const v16bf*)(afrag + (size_t)cc * 512);
                v16bf b0 = *(const v16bf*)(&smem[bCol0 + cc * 32]);
                v16bf b1 = *(const v16bf*)(&smem[bCol1 + cc * 32]);
                acc0 = __builtin_amdgcn_wmma_f32_16x16x32_bf16(
                    false, a, false, b0, (short)0, acc0, false, false);
                acc1 = __builtin_amdgcn_wmma_f32_16x16x32_bf16(
                    false, a, false, b1, (short)0, acc1, false, false);
            }
        }
    }

    // D 16x16 f32 layout: lane<16 -> n=lane, rows 0..7; lane>=16 -> rows 8..15
    const int coBase = coT * 16 + ((lane < 16) ? 0 : 8);
    for (int sub = 0; sub < 2; ++sub) {
        const int xIdx = g * 32 + sub * 16 + nL;
        const int ox = px ? (1 + 2 * xIdx) : (2 * xIdx);
        if (ox < OH) {
            const v8f& a = sub ? acc1 : acc0;
            for (int r = 0; r < 8; ++r) {
                int co = coBase + r;
                convo[(((size_t)b * COUT + co) * OH + oy) * OH + ox] = a[r];
            }
        }
    }
}

// ---------------------------------------------------------------------------
// 4) blur: separable [1,3,3,1]/4 per axis, pad (1,1) over 129x129 -> 128x128
// ---------------------------------------------------------------------------
__global__ void blur_kernel(const float* __restrict__ convo,
                            float* __restrict__ out, size_t total) {
    size_t idx = (size_t)blockIdx.x * blockDim.x + threadIdx.x;
    if (idx >= total) return;
    int fx = (int)(idx & 127);
    int fy = (int)((idx >> 7) & 127);
    size_t pl = idx >> 14;                   // b*COUT + c plane
    const float* p = convo + pl * (size_t)OH * OH;
    const float tap[4] = {0.25f, 0.75f, 0.75f, 0.25f};
    float sum = 0.f;
    for (int ty = 0; ty < 4; ++ty) {
        int sy = fy - 1 + ty;
        if (sy < 0 || sy >= OH) continue;
        float rs = 0.f;
        for (int tx = 0; tx < 4; ++tx) {
            int sx = fx - 1 + tx;
            if (sx < 0 || sx >= OH) continue;
            rs += tap[tx] * p[(size_t)sy * OH + sx];
        }
        sum += tap[ty] * rs;
    }
    out[idx] = sum;
}

// ---------------------------------------------------------------------------
extern "C" void kernel_launch(void* const* d_in, const int* in_sizes, int n_in,
                              void* d_out, int out_size, void* d_ws, size_t ws_size,
                              hipStream_t stream) {
    (void)in_sizes; (void)n_in; (void)out_size; (void)ws_size;
    const float* input  = (const float*)d_in[0]; // [8,512,64,64]
    const float* style  = (const float*)d_in[1]; // [8,512]
    const float* weight = (const float*)d_in[2]; // [1,256,512,3,3]
    const float* modw   = (const float*)d_in[3]; // [512,512]
    const float* modb   = (const float*)d_in[4]; // [512]
    float* out = (float*)d_out;                  // [8,256,128,128]

    char* ws = (char*)d_ws;
    float*          s     = (float*)ws;                              // 16 KB
    unsigned short* aw    = (unsigned short*)(ws + 16384);           // 18 MB bf16 A-frags
    float*          convo = (float*)(ws + 16384 + 18874368);         // 136 MB fp32

    mod_kernel<<<(NB * CIN + 127) / 128, 128, 0, stream>>>(style, modw, modb, s);
    wprep_kernel<<<NB * COUT, 256, 0, stream>>>(weight, s, aw);

    dim3 grid(5, OH, NB * 2);
    convt_wmma_kernel<<<grid, 256, 0, stream>>>(input, aw, convo);

    size_t total = (size_t)NB * COUT * FH * FH;
    blur_kernel<<<(unsigned)((total + 255) / 256), 256, 0, stream>>>(convo, out, total);
}
